// DeformableSelfAttention_25426206392516
// MI455X (gfx1250) — matrix operand order repaired and checked by reading, not compiled, not run on hardware
//
#include <hip/hip_runtime.h>
#include <hip/hip_bf16.h>

#define NHEAD 8
#define NPNT 4
#define DIM 256
#define BATCH 4
#define HH 96
#define WW 96
#define HW (HH * WW)          // 9216
#define CONV_M 96             // 64 offset channels + 32 aw channels
#define CONV_K (9 * DIM)      // 2304

typedef _Float16 f16;
typedef __attribute__((ext_vector_type(16))) _Float16 v16h;
typedef __attribute__((ext_vector_type(8)))  float    v8f;
typedef __attribute__((address_space(3))) f16 lds_f16;

union FragU { v16h v; uint4 q[2]; };

// ---------------------------------------------------------------------------
// Pack: fp32 -> f16 conversions + weight repack (conv weights to [m][tap*256+c])
// ---------------------------------------------------------------------------
__global__ __launch_bounds__(256) void pack_kernel(
    const float* __restrict__ x,
    const float* __restrict__ kv_w, const float* __restrict__ kv_b,
    const float* __restrict__ off_w, const float* __restrict__ off_b,
    const float* __restrict__ aw_w,  const float* __restrict__ aw_b,
    const float* __restrict__ out_w, const float* __restrict__ out_b,
    f16* __restrict__ xb, f16* __restrict__ wv, f16* __restrict__ wo,
    f16* __restrict__ wconv, float* __restrict__ vbias,
    float* __restrict__ cbias, float* __restrict__ obias)
{
    int i = blockIdx.x * 256 + threadIdx.x;
    const int NXB = BATCH * DIM * HW;
    if (i < NXB) { xb[i] = (f16)x[i]; return; }
    i -= NXB;
    if (i < DIM * DIM) { wv[i] = (f16)kv_w[DIM * DIM + i]; return; }  // V half of kv_w
    i -= DIM * DIM;
    if (i < DIM * DIM) { wo[i] = (f16)out_w[i]; return; }
    i -= DIM * DIM;
    if (i < CONV_M * CONV_K) {
        int m = i / CONV_K;
        int r = i - m * CONV_K;
        int tap = r / DIM;          // kh*3+kw
        int c   = r - tap * DIM;
        float v = (m < 64) ? off_w[(size_t)m * CONV_K + c * 9 + tap]
                           : aw_w[(size_t)(m - 64) * CONV_K + c * 9 + tap];
        wconv[i] = (f16)v;
        return;
    }
    i -= CONV_M * CONV_K;
    if (i < DIM)   { vbias[i] = kv_b[DIM + i]; return; }
    i -= DIM;
    if (i < CONV_M){ cbias[i] = (i < 64) ? off_b[i] : aw_b[i - 64]; return; }
    i -= CONV_M;
    if (i < DIM)   { obias[i] = out_b[i]; return; }
}

// ---------------------------------------------------------------------------
// 256x256 GEMM over N=HW per batch:  Y[b,m,n] = sum_k W[m,k] * X[b,k,n] + bias
// Block: 8 waves; block tile M=64 x N=128; wave tile 16 x 64 (4 accumulators).
//  - W block (64x256, 32 KB) copied global->LDS once via async-LDS DMA
//    (global_load_async_to_lds_b128 / s_wait_asynccnt): W is K-contiguous, so
//    the A fragment needs no transpose and reads are ds_load_b128 pairs.
//  - X tile (32K x 128N) double-buffered in LDS with a VGPR transpose to the
//    WMMA 16-bit B(32x16) lane layout; staging of step t+1 overlaps WMMAs of t.
// ---------------------------------------------------------------------------
__global__ __launch_bounds__(256) void gemm_kernel(
    const f16* __restrict__ Wm,     // [256][256] row-major (M,K)
    const f16* __restrict__ Xm,     // [B][256][HW]
    const float* __restrict__ bias, // [256]
    float* __restrict__ Ym)         // [B][256][HW]
{
    __shared__ f16 tileA[64 * 256];      // 32 KB: W rows mb..mb+63
    __shared__ f16 tileB[2][128 * 32];   // 2 x 8 KB, [n][k] (k contiguous)

    const int b    = blockIdx.z;
    const int nb   = blockIdx.x * 128;
    const int mb   = blockIdx.y * 64;
    const int tid  = threadIdx.x;
    const int wave = tid >> 5;
    const int lane = tid & 31;
    const int lhalf = lane >> 4;         // 0: lanes 0-15, 1: lanes 16-31
    const int l16   = lane & 15;
    const int mw   = wave >> 1;          // 0..3 (m-wave)
    const int nh   = (wave & 1) * 64;    // n-half within block tile
    const int m0l  = mw * 16;            // local m offset

    const f16* Xb = Xm + (size_t)b * DIM * HW;

    // ---- async DMA: W block (64x256 f16 = 32 KB) straight into LDS ----
    {
        const f16* wsrc = Wm + (size_t)mb * DIM;
        const unsigned lbase = (unsigned)(size_t)(lds_f16*)tileA;
#pragma unroll
        for (int i = 0; i < 8; ++i) {    // 32 KB / (256 thr * 16 B) = 8 chunks
            unsigned loff = lbase + (unsigned)(tid * 16 + i * 4096);
            unsigned long long gaddr =
                (unsigned long long)(size_t)wsrc + (unsigned)(tid * 16 + i * 4096);
            asm volatile("global_load_async_to_lds_b128 %0, %1, off"
                         :: "v"(loff), "v"(gaddr) : "memory");
        }
        asm volatile("s_wait_asynccnt 0x0" ::: "memory");
    }

    // ---- B staging: 32(K) x 128(N) slab, transposed into [n][k] ----
    auto stageB = [&](int kt, int buf) {
        const int kk = tid >> 3;         // 0..31 K row
        const int ch = tid & 7;          // 0..7  -> 16-element N chunk
        const f16* gp = Xb + (size_t)(kt * 32 + kk) * HW + nb + ch * 16;
        __builtin_prefetch(gp + 32 * HW, 0, 0);       // next K slab
        const uint4 q0 = *(const uint4*)gp;
        const uint4 q1 = *(const uint4*)(gp + 8);
        const f16* r0 = (const f16*)&q0;
        const f16* r1 = (const f16*)&q1;
        f16* dst = &tileB[buf][0];
#pragma unroll
        for (int j = 0; j < 8; ++j) {
            dst[(ch * 16 + j)     * 32 + kk] = r0[j];
            dst[(ch * 16 + 8 + j) * 32 + kk] = r1[j];
        }
    };

    v8f acc[4] = {};
    stageB(0, 0);
    __syncthreads();

    for (int kt = 0; kt < 8; ++kt) {
        const int cur = kt & 1;
        if (kt < 7) stageB(kt + 1, cur ^ 1);   // overlap staging with WMMAs

        const int k0 = kt * 32;
        // A fragment from LDS (16-bit A 16x32 lane layout; K-contiguous rows)
        FragU a;
        const f16* ap = &tileA[(m0l + l16) * DIM + k0 + lhalf * 8];
        a.q[0] = *(const uint4*)(ap);          // K 0..7   (or 8..15)
        a.q[1] = *(const uint4*)(ap + 16);     // K 16..23 (or 24..31)
#pragma unroll
        for (int j = 0; j < 4; ++j) {
            FragU bf;
            const f16* bp = &tileB[cur][(nh + j * 16 + l16) * 32 + lhalf * 8];
            bf.q[0] = *(const uint4*)(bp);
            bf.q[1] = *(const uint4*)(bp + 16);
            acc[j] = __builtin_amdgcn_wmma_f32_16x16x32_f16(
                false, a.v, false, bf.v, (short)0, acc[j], false, false);
        }
        __syncthreads();
    }

    // epilogue: C/D 16x16 f32 layout (VGPR r: lanes0-15 M=r, lanes16-31 M=8+r)
#pragma unroll
    for (int j = 0; j < 4; ++j)
#pragma unroll
        for (int r = 0; r < 8; ++r) {
            const int m = mb + m0l + lhalf * 8 + r;
            const int n = nb + nh + j * 16 + l16;
            Ym[(size_t)b * DIM * HW + (size_t)m * HW + n] = acc[j][r] + bias[m];
        }
}

// ---------------------------------------------------------------------------
// 3x3 conv as implicit GEMM: M=96 channels, K=2304, N=96 (one image row/block)
// Block: 6 waves; wave tile 16 x 96 (6 accumulators). Zero-padded row slices
// staged to LDS per (tap, c-chunk).
// ---------------------------------------------------------------------------
__global__ __launch_bounds__(192) void conv_kernel(
    const f16* __restrict__ Wc,      // [96][2304]
    const f16* __restrict__ Xb,      // [B][256][HW]
    const float* __restrict__ cbias, // [96]
    float* __restrict__ Yc)          // [B][96][HW]
{
    __shared__ f16 tileB[96 * 32];   // [n][k]
    const int y    = blockIdx.x;     // image row
    const int b    = blockIdx.y;
    const int tid  = threadIdx.x;
    const int wave = tid >> 5;       // 0..5
    const int lane = tid & 31;
    const int lhalf = lane >> 4;
    const int l16   = lane & 15;
    const int m0    = wave * 16;
    const f16* Xp = Xb + (size_t)b * DIM * HW;

    v8f acc[6] = {};
    for (int kt = 0; kt < 72; ++kt) {
        const int tap = kt >> 3;                  // 0..8
        const int dy = tap / 3 - 1, dx = tap % 3 - 1;
        const int c0 = (kt & 7) * 32;
        const int sy = y + dy;
        const bool rowok = (sy >= 0) && (sy < HH);
        // stage 32(K) x 96(N) zero-padded slice
#pragma unroll
        for (int e = 0; e < 16; ++e) {
            const int idx = e * 192 + tid;        // 0..3071
            const int kk = idx / 96;
            const int nn = idx - kk * 96;
            const int sx = nn + dx;
            f16 val = (f16)0.0f;
            if (rowok && sx >= 0 && sx < WW)
                val = Xp[(size_t)(c0 + kk) * HW + sy * WW + sx];
            tileB[nn * 32 + kk] = val;
        }
        __syncthreads();

        FragU a;
        const f16* ap = Wc + (size_t)(m0 + l16) * CONV_K + kt * 32 + lhalf * 8;
        a.q[0] = *(const uint4*)(ap);
        a.q[1] = *(const uint4*)(ap + 16);
#pragma unroll
        for (int j = 0; j < 6; ++j) {
            FragU bf;
            const f16* bp = &tileB[(j * 16 + l16) * 32 + lhalf * 8];
            bf.q[0] = *(const uint4*)(bp);
            bf.q[1] = *(const uint4*)(bp + 16);
            acc[j] = __builtin_amdgcn_wmma_f32_16x16x32_f16(
                false, a.v, false, bf.v, (short)0, acc[j], false, false);
        }
        __syncthreads();
    }

#pragma unroll
    for (int j = 0; j < 6; ++j)
#pragma unroll
        for (int r = 0; r < 8; ++r) {
            const int m = m0 + lhalf * 8 + r;
            const int n = j * 16 + l16;
            Yc[((size_t)b * CONV_M + m) * HW + y * WW + n] = acc[j][r] + cbias[m];
        }
}

// ---------------------------------------------------------------------------
// Softmax over points + bilinear sampling of V, weighted sum -> S (f16)
// One thread per (b, head, pixel); inner loop over 32 head-dims.
// ---------------------------------------------------------------------------
__global__ __launch_bounds__(256) void sample_kernel(
    const float* __restrict__ conv,  // [B][96][HW]
    const float* __restrict__ V,     // [B][256][HW]
    f16* __restrict__ S)             // [B][256][HW]
{
    const int gid = blockIdx.x * 256 + threadIdx.x;
    if (gid >= BATCH * NHEAD * HW) return;
    const int s  = gid % HW;
    const int bh = gid / HW;
    const int h  = bh % NHEAD;
    const int b  = bh / NHEAD;
    const int y  = s / WW;
    const int x  = s - y * WW;

    const float* cp = conv + (size_t)b * CONV_M * HW;

    // softmax over the 4 sampling points
    float lg[NPNT];
#pragma unroll
    for (int p = 0; p < NPNT; ++p)
        lg[p] = cp[(size_t)(64 + h * NPNT + p) * HW + s];
    float mx = fmaxf(fmaxf(lg[0], lg[1]), fmaxf(lg[2], lg[3]));
    float sum = 0.f;
#pragma unroll
    for (int p = 0; p < NPNT; ++p) { lg[p] = expf(lg[p] - mx); sum += lg[p]; }
    const float inv = 1.0f / sum;

    float wgt[16];
    int   idxs[16];
    const float SC = 95.0f / 96.0f;   // (W-1)/W grid scale
#pragma unroll
    for (int p = 0; p < NPNT; ++p) {
        const float offx = cp[(size_t)(((h * NPNT + p) * 2) + 0) * HW + s];
        const float offy = cp[(size_t)(((h * NPNT + p) * 2) + 1) * HW + s];
        const float px = (float)x + offx * SC;
        const float py = (float)y + offy * SC;
        const float x0f = floorf(px), y0f = floorf(py);
        const float fx = px - x0f, fy = py - y0f;
        const float aw = lg[p] * inv;
        const int xi0 = (int)x0f, yi0 = (int)y0f;
        const int xi1 = xi0 + 1,  yi1 = yi0 + 1;
        const bool vx0 = xi0 >= 0 && xi0 < WW, vx1 = xi1 >= 0 && xi1 < WW;
        const bool vy0 = yi0 >= 0 && yi0 < HH, vy1 = yi1 >= 0 && yi1 < HH;
        const int cx0 = min(max(xi0, 0), WW - 1), cx1 = min(max(xi1, 0), WW - 1);
        const int cy0 = min(max(yi0, 0), HH - 1), cy1 = min(max(yi1, 0), HH - 1);
        wgt[p*4+0] = (vx0 && vy0) ? (1.f-fx)*(1.f-fy)*aw : 0.f; idxs[p*4+0] = cy0*WW+cx0;
        wgt[p*4+1] = (vx1 && vy0) ? fx*(1.f-fy)*aw : 0.f;       idxs[p*4+1] = cy0*WW+cx1;
        wgt[p*4+2] = (vx0 && vy1) ? (1.f-fx)*fy*aw : 0.f;       idxs[p*4+2] = cy1*WW+cx0;
        wgt[p*4+3] = (vx1 && vy1) ? fx*fy*aw : 0.f;             idxs[p*4+3] = cy1*WW+cx1;
    }

#pragma unroll 4
    for (int d = 0; d < 32; ++d) {
        const float* vp = V + ((size_t)b * DIM + h * 32 + d) * HW;
        float accv = 0.f;
#pragma unroll
        for (int i = 0; i < 16; ++i)
            accv = fmaf(wgt[i], vp[idxs[i]], accv);
        S[((size_t)b * DIM + h * 32 + d) * HW + s] = (f16)accv;
    }
}

// ---------------------------------------------------------------------------
extern "C" void kernel_launch(void* const* d_in, const int* in_sizes, int n_in,
                              void* d_out, int out_size, void* d_ws, size_t ws_size,
                              hipStream_t stream)
{
    const float* x     = (const float*)d_in[0];
    // d_in[1], d_in[2] (q_w, q_b) are unused by the reference
    const float* kv_w  = (const float*)d_in[3];
    const float* kv_b  = (const float*)d_in[4];
    const float* off_w = (const float*)d_in[5];
    const float* off_b = (const float*)d_in[6];
    const float* aw_w  = (const float*)d_in[7];
    const float* aw_b  = (const float*)d_in[8];
    const float* out_w = (const float*)d_in[9];
    const float* out_b = (const float*)d_in[10];

    char* ws = (char*)d_ws;
    size_t off = 0;
    f16*   XB   = (f16*)(ws + off);   off += (size_t)BATCH * DIM * HW * 2;     // x in f16
    float* Vbuf = (float*)(ws + off); off += (size_t)BATCH * DIM * HW * 4;     // V projection
    float* CONV = (float*)(ws + off); off += (size_t)BATCH * CONV_M * HW * 4;  // offsets+aw
    f16*   Sbuf = (f16*)(ws + off);   off += (size_t)BATCH * DIM * HW * 2;     // sampled (f16)
    f16*   WV   = (f16*)(ws + off);   off += (size_t)DIM * DIM * 2;
    f16*   WO   = (f16*)(ws + off);   off += (size_t)DIM * DIM * 2;
    f16*   WC   = (f16*)(ws + off);   off += (size_t)CONV_M * CONV_K * 2;
    float* VB   = (float*)(ws + off); off += 1024;
    float* CB   = (float*)(ws + off); off += 512;
    float* OB   = (float*)(ws + off); off += 1024;

    const int ntot = BATCH * DIM * HW + DIM * DIM + DIM * DIM + CONV_M * CONV_K
                   + DIM + CONV_M + DIM;
    pack_kernel<<<(ntot + 255) / 256, 256, 0, stream>>>(
        x, kv_w, kv_b, off_w, off_b, aw_w, aw_b, out_w, out_b,
        XB, WV, WO, WC, VB, CB, OB);

    // V = Wv * X + vb
    gemm_kernel<<<dim3(HW / 128, DIM / 64, BATCH), 256, 0, stream>>>(WV, XB, VB, Vbuf);
    // offsets + aw logits
    conv_kernel<<<dim3(HH, BATCH), 192, 0, stream>>>(WC, XB, CB, CONV);
    // softmax + bilinear sampling
    sample_kernel<<<(BATCH * NHEAD * HW + 255) / 256, 256, 0, stream>>>(CONV, Vbuf, Sbuf);
    // out = Wo * S + ob
    gemm_kernel<<<dim3(HW / 128, DIM / 64, BATCH), 256, 0, stream>>>(WO, Sbuf, OB, (float*)d_out);
}
